// GNNLayer_58042188038247
// MI455X (gfx1250) — matrix operand order repaired and checked by reading, not compiled, not run on hardware
//
#include <hip/hip_runtime.h>

typedef __attribute__((ext_vector_type(16))) __bf16 v16bf;
typedef __attribute__((ext_vector_type(8)))  float  v8f;

#define DFEAT 256
#define BN_EPS 1e-5f

// ---------------- zero scratch ----------------
__global__ void zero_f32_kernel(float* __restrict__ p, int n) {
    int i = blockIdx.x * blockDim.x + threadIdx.x;
    if (i < n) p[i] = 0.0f;
}

// ---------------- weights f32 -> bf16 ----------------
__global__ void cvt_w_kernel(const float* __restrict__ W1, const float* __restrict__ W2,
                             __bf16* __restrict__ W1b, __bf16* __restrict__ W2b, int n) {
    int i = blockIdx.x * blockDim.x + threadIdx.x;
    if (i < n) {
        W1b[i] = (__bf16)W1[i];
        W2b[i] = (__bf16)W2[i];
    }
}

// ---------------- edge scatter: agg[dst] += h[src]*e, deg[dst] += 1 ----------------
// one wave32 per edge; lane owns 8 contiguous features (2x float4 loads)
__global__ void scatter_kernel(const float* __restrict__ h, const float* __restrict__ e,
                               const int* __restrict__ src, const int* __restrict__ dst,
                               float* __restrict__ agg, float* __restrict__ deg, int nEdges) {
    int gid  = blockIdx.x * blockDim.x + threadIdx.x;
    int edge = gid >> 5;
    int lane = gid & 31;
    if (edge >= nEdges) return;
    int   s  = src[edge];
    int   d  = dst[edge];
    float ev = e[edge];
    const float4* hs = (const float4*)(h + (size_t)s * DFEAT);
    float* ab = agg + (size_t)d * DFEAT + lane * 8;
#pragma unroll
    for (int i = 0; i < 2; ++i) {
        float4 v = hs[lane * 2 + i];
        atomicAdd(ab + i * 4 + 0, v.x * ev);
        atomicAdd(ab + i * 4 + 1, v.y * ev);
        atomicAdd(ab + i * 4 + 2, v.z * ev);
        atomicAdd(ab + i * 4 + 3, v.w * ev);
    }
    if (lane == 0) atomicAdd(deg + d, 1.0f);
}

// ---------------- mean normalize (in place) + bf16 cast ----------------
__global__ void mean_cvt_kernel(float* __restrict__ hn, const float* __restrict__ deg,
                                __bf16* __restrict__ hnb, int total) {
    int i = blockIdx.x * blockDim.x + threadIdx.x;
    if (i >= total) return;
    int   n  = i >> 8;              // DFEAT == 256
    float dg = deg[n];
    float v  = (dg > 0.0f) ? hn[i] / dg : 0.0f;
    hn[i]  = v;
    hnb[i] = (__bf16)v;
}

// ---------------- WMMA GEMM: out = relu(A @ W^T + bias) ----------------
// A: [M,256] bf16 row-major.  W: [256,256] bf16 row-major (out,in) -> B[k,n]=W[n,k].
// One wave per 16x16 output tile; K=256 -> 8x v_wmma_f32_16x16x32_bf16.
// A fragment (ISA 7.12.2, 16-bit A 16x32): lane l: row = l&15, half = l>>4,
//   elems 0..7  <- K = kb + half*8 .. +7        (one b128)
//   elems 8..15 <- K = kb + 16 + half*8 .. +7   (one b128)
// B fragment (32x16): lane l: col n = l&15, same K order as A (two b128)
// C/D (f32 16x16): lane l holds col n = l&15, VGPR r -> row = r + 8*half.
__global__ void gemm_wmma_kernel(const __bf16* __restrict__ A, const __bf16* __restrict__ W,
                                 const float* __restrict__ bias,
                                 float* __restrict__ outF, __bf16* __restrict__ outB,
                                 int tilesM) {
    int wave = (blockIdx.x * blockDim.x + threadIdx.x) >> 5;
    int lane = threadIdx.x & 31;
    int tn   = wave & 15;      // 16 N-tiles (256/16)
    int tm   = wave >> 4;
    if (tm >= tilesM) return;  // wave-uniform

    int half = lane >> 4;
    int l15  = lane & 15;
    int m_row = tm * 16 + l15;
    int n_col = tn * 16 + l15;

    const __bf16* arow = A + (size_t)m_row * DFEAT;
    const __bf16* brow = W + (size_t)n_col * DFEAT;

    v8f acc = {};
#pragma unroll
    for (int kb = 0; kb < DFEAT; kb += 32) {
        v16bf a, b;
        *((uint4*)&a)     = *(const uint4*)(arow + kb + half * 8);
        *((uint4*)&a + 1) = *(const uint4*)(arow + kb + 16 + half * 8);
        const __bf16* bp  = brow + kb + half * 8;
        *((uint4*)&b)     = *(const uint4*)(bp);
        *((uint4*)&b + 1) = *(const uint4*)(bp + 16);
        acc = __builtin_amdgcn_wmma_f32_16x16x32_bf16(
            /*neg_a=*/false, a, /*neg_b=*/false, b,
            /*c_mod=*/(short)0, acc, /*reuse_a=*/false, /*reuse_b=*/false);
    }

    float bv      = bias[n_col];
    int   rowBase = tm * 16 + half * 8;
#pragma unroll
    for (int r = 0; r < 8; ++r) {
        float v = acc[r] + bv;
        v = (v > 0.0f) ? v : 0.0f;
        size_t off = (size_t)(rowBase + r) * DFEAT + n_col;
        if (outF) outF[off] = v;
        if (outB) outB[off] = (__bf16)v;
    }
}

// ---------------- BN statistics: per-feature sum & sumsq ----------------
// blockDim = 256 (thread = feature, coalesced rows); block covers a node range.
__global__ void bn_stats_kernel(const float* __restrict__ x, float* __restrict__ stats,
                                int nNodes) {
    int f   = threadIdx.x;
    int per = (nNodes + gridDim.x - 1) / gridDim.x;
    int s0  = blockIdx.x * per;
    int s1  = s0 + per; if (s1 > nNodes) s1 = nNodes;
    float s = 0.0f, ss = 0.0f;
    for (int n = s0; n < s1; ++n) {
        float v = x[(size_t)n * DFEAT + f];
        s  += v;
        ss += v * v;
    }
    atomicAdd(&stats[f], s);
    atomicAdd(&stats[DFEAT + f], ss);
}

// ---------------- BN apply + residual ----------------
__global__ void bn_final_kernel(const float* __restrict__ x, const float* __restrict__ hn,
                                const float* __restrict__ stats,
                                const float* __restrict__ gamma, const float* __restrict__ beta,
                                float* __restrict__ out, int total, float invN) {
    int i = blockIdx.x * blockDim.x + threadIdx.x;
    if (i >= total) return;
    int   f   = i & (DFEAT - 1);
    float mu  = stats[f] * invN;
    float var = stats[DFEAT + f] * invN - mu * mu;
    out[i] = gamma[f] * (x[i] - mu) * __frsqrt_rn(var + BN_EPS) + beta[f] + hn[i];
}

extern "C" void kernel_launch(void* const* d_in, const int* in_sizes, int n_in,
                              void* d_out, int out_size, void* d_ws, size_t ws_size,
                              hipStream_t stream) {
    const float* h     = (const float*)d_in[0];
    const float* e     = (const float*)d_in[1];
    const int*   src   = (const int*)  d_in[2];
    const int*   dst   = (const int*)  d_in[3];
    const float* W1    = (const float*)d_in[4];
    const float* b1    = (const float*)d_in[5];
    const float* W2    = (const float*)d_in[6];
    const float* b2    = (const float*)d_in[7];
    const float* gamma = (const float*)d_in[8];
    const float* beta  = (const float*)d_in[9];
    float* out = (float*)d_out;

    const int    nNodes = in_sizes[0] / DFEAT;   // 50000
    const int    nEdges = in_sizes[2];           // 800000
    const size_t feat   = (size_t)nNodes * DFEAT;

    // workspace layout (agg|deg|stats contiguous so a single zero pass covers them)
    char* ws = (char*)d_ws;
    float*  agg   = (float*)ws;  ws += feat * 4;                 // becomes hn after mean
    float*  deg   = (float*)ws;  ws += (size_t)nNodes * 4;
    float*  stats = (float*)ws;  ws += 2 * DFEAT * 4;
    float*  x2    = (float*)ws;  ws += feat * 4;
    __bf16* hnb   = (__bf16*)ws; ws += feat * 2;
    __bf16* x1b   = (__bf16*)ws; ws += feat * 2;
    __bf16* W1b   = (__bf16*)ws; ws += (size_t)DFEAT * DFEAT * 2;
    __bf16* W2b   = (__bf16*)ws; ws += (size_t)DFEAT * DFEAT * 2;

    // 1) zero accumulators (agg + deg + stats are contiguous floats)
    int zn = (int)(feat + nNodes + 2 * DFEAT);
    zero_f32_kernel<<<(zn + 255) / 256, 256, 0, stream>>>(agg, zn);

    // 2) weights -> bf16
    int wn = DFEAT * DFEAT;
    cvt_w_kernel<<<(wn + 255) / 256, 256, 0, stream>>>(W1, W2, W1b, W2b, wn);

    // 3) edge scatter (one wave per edge)
    long long sthreads = (long long)nEdges * 32;
    scatter_kernel<<<(int)((sthreads + 255) / 256), 256, 0, stream>>>(
        h, e, src, dst, agg, deg, nEdges);

    // 4) mean + bf16 cast (agg -> hn in place)
    mean_cvt_kernel<<<(int)((feat + 255) / 256), 256, 0, stream>>>(agg, deg, hnb, (int)feat);

    // 5) GEMM1: x1 = relu(hn @ W1^T + b1)  -> bf16 only
    int tilesM = (nNodes + 15) / 16;             // 3125
    int waves  = tilesM * 16;                    // 50000 tiles
    int blocks = (waves + 7) / 8;                // 8 waves / block
    gemm_wmma_kernel<<<blocks, 256, 0, stream>>>(hnb, W1b, b1, nullptr, x1b, tilesM);

    // 6) GEMM2: x2 = relu(x1 @ W2^T + b2)  -> f32 only
    gemm_wmma_kernel<<<blocks, 256, 0, stream>>>(x1b, W2b, b2, x2, nullptr, tilesM);

    // 7) BN stats
    bn_stats_kernel<<<256, 256, 0, stream>>>(x2, stats, nNodes);

    // 8) BN apply + residual
    bn_final_kernel<<<(int)((feat + 255) / 256), 256, 0, stream>>>(
        x2, agg, stats, gamma, beta, out, (int)feat, 1.0f / (float)nNodes);
}
